// MaskMLP_61452392071889
// MI455X (gfx1250) — compile-verified, matrix operand order checked by reading, not run on hardware
//
#include <hip/hip_runtime.h>
#include <math.h>

typedef __attribute__((ext_vector_type(16))) __bf16 v16bf;
typedef __attribute__((ext_vector_type(8)))  float  v8f;

#define B_SZ   1024
#define D_SZ   4096
#define M_MOD  128
#define N0     4096   /* = M_MOD * 32 */

#if defined(__has_builtin)
#if __has_builtin(__builtin_amdgcn_tensor_load_to_lds) && __has_builtin(__builtin_amdgcn_s_wait_tensorcnt)
#define HAS_TDM 1
#endif
#endif
#ifndef HAS_TDM
#define HAS_TDM 0
#endif

// ---------------- bf16 split helpers (branch-free RNE, storage-only bf16) ----
__device__ __forceinline__ unsigned short f2bf(float x) {
  unsigned u = __float_as_uint(x);
  u += 0x7FFFu + ((u >> 16) & 1u);        // round-to-nearest-even
  return (unsigned short)(u >> 16);
}
__device__ __forceinline__ float bf2f(unsigned short h) {
  return __uint_as_float((unsigned)h << 16);
}
__device__ __forceinline__ void f32_split(float x, unsigned short& h, unsigned short& l) {
  h = f2bf(x);
  l = f2bf(x - bf2f(h));
}

// bf16x3 WMMA term: D = A(16x32 bf16) * B(32x16 bf16) + C(f32)
__device__ __forceinline__ v8f wmma_bf16(v16bf a, v16bf b, v8f c) {
  return __builtin_amdgcn_wmma_f32_16x16x32_bf16(false, a, false, b, (short)0, c, false, false);
}

// Fragment gather: per-lane K runs [8*half, +7] and [16+8*half, +7] (ISA 16-bit A layout).
__device__ __forceinline__ v16bf load_frag(const unsigned short* p, int half) {
  v16bf f;
  ((float4*)&f)[0] = *(const float4*)(p + 8 * half);
  ((float4*)&f)[1] = *(const float4*)(p + 16 + 8 * half);
  return f;
}

#if HAS_TDM
// ---------------- Tensor Data Mover: 2D tile load, data_size = 2 bytes -------
// D# per CDNA5 ISA ch.8: group0 = {flags/count, lds_addr, global_addr, type=2},
// group1 = {mask/data_size, tensor_dim0/1, tile_dim0/1/2, dim0_stride}, groups
// 2/3 zero (<=2D tensor: tile_dim3/4 = 0 -> unused).
typedef unsigned int u32x4 __attribute__((ext_vector_type(4)));
typedef int          i32x4 __attribute__((ext_vector_type(4)));
typedef int          i32x8 __attribute__((ext_vector_type(8)));

__device__ __forceinline__ void tdm_load_2d_u16(unsigned lds_addr, const void* gaddr,
                                                unsigned tile_d0, unsigned tile_d1,
                                                unsigned tensor_d0, unsigned tensor_d1,
                                                unsigned stride_d0 /* elements */) {
  unsigned long long ga = (unsigned long long)gaddr;
  u32x4 g0;
  g0[0] = 1u;                                   // count=1, user D#, no gather
  g0[1] = lds_addr;                             // LDS byte address (addr[31:0])
  g0[2] = (unsigned)ga;                         // global_addr[31:0]
  g0[3] = (unsigned)(ga >> 32) | (2u << 30);    // global_addr[56:32] | type=2
  i32x8 g1;
  g1[0] = (int)(1u << 16);                      // wg_mask=0, data_size=1 (2B), no flags
  g1[1] = (int)((tensor_d0 & 0xFFFFu) << 16);               // tensor_dim0[15:0]
  g1[2] = (int)((tensor_d0 >> 16) | ((tensor_d1 & 0xFFFFu) << 16));
  g1[3] = (int)((tensor_d1 >> 16) | (tile_d0 << 16));       // tile_dim0
  g1[4] = (int)(tile_d1 & 0xFFFFu);                         // tile_dim1, tile_dim2=0
  g1[5] = (int)stride_d0;                                   // tensor_dim0_stride[31:0]
  g1[6] = 0;
  g1[7] = 0;
  i32x4 z4 = {0, 0, 0, 0};
#if __clang_major__ >= 23
  i32x8 z8 = {0, 0, 0, 0, 0, 0, 0, 0};
  __builtin_amdgcn_tensor_load_to_lds(g0, g1, z4, z4, z8, 0);
#else
  __builtin_amdgcn_tensor_load_to_lds(g0, g1, z4, z4, 0);
#endif
}
#endif // HAS_TDM

// =====================================================================
// Pre-pass: split f32 array into bf16 hi/lo planes (memory-bound, run once)
// =====================================================================
__global__ __launch_bounds__(256) void split_bf16(const float* __restrict__ src,
                                                  unsigned short* __restrict__ hi,
                                                  unsigned short* __restrict__ lo,
                                                  int n4) {
  int i = blockIdx.x * 256 + threadIdx.x;
  if (i >= n4) return;
  float4 v = ((const float4*)src)[i];
  unsigned short h0, l0, h1, l1, h2, l2, h3, l3;
  f32_split(v.x, h0, l0);
  f32_split(v.y, h1, l1);
  f32_split(v.z, h2, l2);
  f32_split(v.w, h3, l3);
  *(ushort4*)&hi[(size_t)i * 4] = make_ushort4(h0, h1, h2, h3);
  *(ushort4*)&lo[(size_t)i * 4] = make_ushort4(l0, l1, l2, l3);
}

// =====================================================================
// Kernel A: H0 = relu(x @ W0^T + b0), bf16x3. Tile 128x128, Kt=32.
// TDM double-buffered staging: wave0 DMA's 4 bf16 planes per chunk into the
// idle LDS buffer while all 8 waves run 24 wmma on the current one.
// =====================================================================
#define TA_M 128
#define TA_N 128
#define TA_K 32

__global__ __launch_bounds__(256) void gemm0_relu_bf16x3(
    const unsigned short* __restrict__ Xh, const unsigned short* __restrict__ Xl,
    const unsigned short* __restrict__ Wh, const unsigned short* __restrict__ Wl,
    const float* __restrict__ b0,
    unsigned short* __restrict__ H0h, unsigned short* __restrict__ H0l) {
#if HAS_TDM
  __shared__ __align__(16) unsigned short xsh[2][TA_M * TA_K];  // 8 KB per plane buf
  __shared__ __align__(16) unsigned short xsl[2][TA_M * TA_K];
  __shared__ __align__(16) unsigned short wsh[2][TA_N * TA_K];
  __shared__ __align__(16) unsigned short wsl[2][TA_N * TA_K];  // total 64 KB
#else
  __shared__ __align__(16) unsigned short xsh[1][TA_M * TA_K];
  __shared__ __align__(16) unsigned short xsl[1][TA_M * TA_K];
  __shared__ __align__(16) unsigned short wsh[1][TA_N * TA_K];
  __shared__ __align__(16) unsigned short wsl[1][TA_N * TA_K];
#endif
  const int n0   = blockIdx.x * TA_N;
  const int m0   = blockIdx.y * TA_M;
  const int tid  = threadIdx.x;
  const int wave = tid >> 5, lane = tid & 31;
  const int l16  = lane & 15, half = lane >> 4;
  const int wm   = wave & 1,  wn   = wave >> 1;

  v8f acc[4][2];
#pragma unroll
  for (int i = 0; i < 4; ++i)
#pragma unroll
    for (int j = 0; j < 2; ++j) acc[i][j] = (v8f){0.f,0.f,0.f,0.f,0.f,0.f,0.f,0.f};

#if HAS_TDM
  // issue first chunk into buffer 0 (one DMA per bf16 plane)
  if (wave == 0) {
    tdm_load_2d_u16((unsigned)(size_t)&xsh[0][0], Xh + (size_t)m0 * D_SZ, TA_K, TA_M, D_SZ, B_SZ, D_SZ);
    tdm_load_2d_u16((unsigned)(size_t)&xsl[0][0], Xl + (size_t)m0 * D_SZ, TA_K, TA_M, D_SZ, B_SZ, D_SZ);
    tdm_load_2d_u16((unsigned)(size_t)&wsh[0][0], Wh + (size_t)n0 * D_SZ, TA_K, TA_N, D_SZ, N0,   D_SZ);
    tdm_load_2d_u16((unsigned)(size_t)&wsl[0][0], Wl + (size_t)n0 * D_SZ, TA_K, TA_N, D_SZ, N0,   D_SZ);
  }
  int buf = 0;
#endif

  for (int kb = 0; kb < D_SZ; kb += TA_K) {
#if HAS_TDM
    if (wave == 0) __builtin_amdgcn_s_wait_tensorcnt(0);  // current buffer landed
    __syncthreads();
    if (wave == 0 && kb + TA_K < D_SZ) {                  // prefetch next chunk
      const int nb = buf ^ 1, kn = kb + TA_K;
      tdm_load_2d_u16((unsigned)(size_t)&xsh[nb][0], Xh + (size_t)m0 * D_SZ + kn, TA_K, TA_M, D_SZ, B_SZ, D_SZ);
      tdm_load_2d_u16((unsigned)(size_t)&xsl[nb][0], Xl + (size_t)m0 * D_SZ + kn, TA_K, TA_M, D_SZ, B_SZ, D_SZ);
      tdm_load_2d_u16((unsigned)(size_t)&wsh[nb][0], Wh + (size_t)n0 * D_SZ + kn, TA_K, TA_N, D_SZ, N0,   D_SZ);
      tdm_load_2d_u16((unsigned)(size_t)&wsl[nb][0], Wl + (size_t)n0 * D_SZ + kn, TA_K, TA_N, D_SZ, N0,   D_SZ);
    }
    const unsigned short* xh_t = xsh[buf];
    const unsigned short* xl_t = xsl[buf];
    const unsigned short* wh_t = wsh[buf];
    const unsigned short* wl_t = wsl[buf];
#else
    // synchronous staging fallback (512 x 16B per plane)
#pragma unroll
    for (int it = 0; it < 2; ++it) {
      int idx = tid + it * 256;
      int row = idx >> 2;
      int c   = (idx & 3) << 3;
      size_t gx = (size_t)(m0 + row) * D_SZ + kb + c;
      size_t gw = (size_t)(n0 + row) * D_SZ + kb + c;
      *(uint4*)&xsh[0][row * TA_K + c] = *(const uint4*)&Xh[gx];
      *(uint4*)&xsl[0][row * TA_K + c] = *(const uint4*)&Xl[gx];
      *(uint4*)&wsh[0][row * TA_K + c] = *(const uint4*)&Wh[gw];
      *(uint4*)&wsl[0][row * TA_K + c] = *(const uint4*)&Wl[gw];
    }
    __syncthreads();
    const unsigned short* xh_t = xsh[0];
    const unsigned short* xl_t = xsl[0];
    const unsigned short* wh_t = wsh[0];
    const unsigned short* wl_t = wsl[0];
#endif

    v16bf ah[4], bh[2], bl[2];
#pragma unroll
    for (int i = 0; i < 4; ++i)
      ah[i] = load_frag(&xh_t[(wm * 64 + i * 16 + l16) * TA_K], half);
#pragma unroll
    for (int j = 0; j < 2; ++j)
      bh[j] = load_frag(&wh_t[(wn * 32 + j * 16 + l16) * TA_K], half);
#pragma unroll
    for (int i = 0; i < 4; ++i)
#pragma unroll
      for (int j = 0; j < 2; ++j) acc[i][j] = wmma_bf16(ah[i], bh[j], acc[i][j]);   // hi*hi
#pragma unroll
    for (int j = 0; j < 2; ++j)
      bl[j] = load_frag(&wl_t[(wn * 32 + j * 16 + l16) * TA_K], half);
#pragma unroll
    for (int i = 0; i < 4; ++i)
#pragma unroll
      for (int j = 0; j < 2; ++j) acc[i][j] = wmma_bf16(ah[i], bl[j], acc[i][j]);   // hi*lo
#pragma unroll
    for (int i = 0; i < 4; ++i) {
      v16bf al = load_frag(&xl_t[(wm * 64 + i * 16 + l16) * TA_K], half);
#pragma unroll
      for (int j = 0; j < 2; ++j) acc[i][j] = wmma_bf16(al, bh[j], acc[i][j]);      // lo*hi
    }
    __syncthreads();
#if HAS_TDM
    buf ^= 1;
#endif
  }

  // epilogue: bias + relu, split to bf16 hi/lo, store
#pragma unroll
  for (int i = 0; i < 4; ++i) {
#pragma unroll
    for (int j = 0; j < 2; ++j) {
      const int col  = n0 + wn * 32 + j * 16 + l16;
      const float bb = b0[col];
      const int rb   = m0 + wm * 64 + i * 16 + half * 8;
#pragma unroll
      for (int e = 0; e < 8; ++e) {
        float v = acc[i][j][e] + bb;
        v = v > 0.f ? v : 0.f;
        unsigned short h, l;
        f32_split(v, h, l);
        H0h[(size_t)(rb + e) * N0 + col] = h;
        H0l[(size_t)(rb + e) * N0 + col] = l;
      }
    }
  }
}

// =====================================================================
// Kernel B: fused per-module layers 1..4 + output dot + sigmoid (bf16x3).
// Activations stay in LDS as bf16 hi/lo; weights stream from L2 (pre-split).
// =====================================================================
#define BT 16

__device__ __forceinline__ void layer_gemm(const unsigned short* __restrict__ inh,
                                           const unsigned short* __restrict__ inl, int in_dim,
                                           unsigned short* __restrict__ outh,
                                           unsigned short* __restrict__ outl, int out_dim,
                                           const unsigned short* __restrict__ Wmh,
                                           const unsigned short* __restrict__ Wml,
                                           const float* __restrict__ bm,
                                           int wave, int l16, int half) {
  const int ntiles = out_dim >> 4;
  for (int nt = wave; nt < ntiles; nt += 8) {
    v8f acc = (v8f){0.f,0.f,0.f,0.f,0.f,0.f,0.f,0.f};
    const unsigned short* arh = inh + (size_t)l16 * in_dim;
    const unsigned short* arl = inl + (size_t)l16 * in_dim;
    const unsigned short* brh = Wmh + (size_t)(nt * 16 + l16) * in_dim;
    const unsigned short* brl = Wml + (size_t)(nt * 16 + l16) * in_dim;
    for (int k = 0; k < in_dim; k += 32) {
      v16bf ah = load_frag(arh + k, half);
      v16bf bh = load_frag(brh + k, half);
      acc = wmma_bf16(ah, bh, acc);
      v16bf bl = load_frag(brl + k, half);
      acc = wmma_bf16(ah, bl, acc);
      v16bf al = load_frag(arl + k, half);
      acc = wmma_bf16(al, bh, acc);
    }
    const int c    = nt * 16 + l16;
    const float bb = bm[c];
    const int rb   = half * 8;
#pragma unroll
    for (int e = 0; e < 8; ++e) {
      float v = acc[e] + bb;
      v = v > 0.f ? v : 0.f;
      unsigned short h, l;
      f32_split(v, h, l);
      outh[(size_t)(rb + e) * out_dim + c] = h;
      outl[(size_t)(rb + e) * out_dim + c] = l;
    }
  }
}

__global__ __launch_bounds__(256) void fused_modules(
    const unsigned short* __restrict__ H0h, const unsigned short* __restrict__ H0l,
    const unsigned short* __restrict__ W1h, const unsigned short* __restrict__ W1l, const float* __restrict__ b1,
    const unsigned short* __restrict__ W2h, const unsigned short* __restrict__ W2l, const float* __restrict__ b2,
    const unsigned short* __restrict__ W3h, const unsigned short* __restrict__ W3l, const float* __restrict__ b3,
    const unsigned short* __restrict__ W4h, const unsigned short* __restrict__ W4l, const float* __restrict__ b4,
    const float* __restrict__ Wout, const float* __restrict__ bout,
    float* __restrict__ out) {
  __shared__ __align__(16) unsigned short b0h[BT * 512];  // 16 KB  (h0:32, h2:128, h4:512)
  __shared__ __align__(16) unsigned short b0l[BT * 512];  // 16 KB
  __shared__ __align__(16) unsigned short b1h_[BT * 256]; //  8 KB  (h1:64,  h3:256)
  __shared__ __align__(16) unsigned short b1l_[BT * 256]; //  8 KB
  const int m    = blockIdx.x;
  const int bb0  = blockIdx.y * BT;
  const int tid  = threadIdx.x;
  const int wave = tid >> 5, lane = tid & 31;
  const int l16  = lane & 15, half = lane >> 4;

  if (tid < 64) {
    int row = tid >> 2;
    int c   = (tid & 3) << 3;
    size_t g = (size_t)(bb0 + row) * N0 + m * 32 + c;
    *(uint4*)&b0h[row * 32 + c] = *(const uint4*)&H0h[g];
    *(uint4*)&b0l[row * 32 + c] = *(const uint4*)&H0l[g];
  }
  __syncthreads();

  layer_gemm(b0h,  b0l,   32, b1h_, b1l_,  64, W1h + (size_t)m *  64 *  32, W1l + (size_t)m *  64 *  32, b1 + m *  64, wave, l16, half);
  __syncthreads();
  layer_gemm(b1h_, b1l_,  64, b0h,  b0l,  128, W2h + (size_t)m * 128 *  64, W2l + (size_t)m * 128 *  64, b2 + m * 128, wave, l16, half);
  __syncthreads();
  layer_gemm(b0h,  b0l,  128, b1h_, b1l_, 256, W3h + (size_t)m * 256 * 128, W3l + (size_t)m * 256 * 128, b3 + m * 256, wave, l16, half);
  __syncthreads();
  layer_gemm(b1h_, b1l_, 256, b0h,  b0l,  512, W4h + (size_t)m * 512 * 256, W4l + (size_t)m * 512 * 256, b4 + m * 512, wave, l16, half);
  __syncthreads();

  {
    const float* wv = Wout + (size_t)m * 512;
    const int row = tid >> 4;        // 0..15
    const int q   = tid & 15;
    float s = 0.f;
    for (int i = q; i < 512; i += 16)
      s += (bf2f(b0h[row * 512 + i]) + bf2f(b0l[row * 512 + i])) * wv[i];
    s += __shfl_xor(s, 8, 32);
    s += __shfl_xor(s, 4, 32);
    s += __shfl_xor(s, 2, 32);
    s += __shfl_xor(s, 1, 32);
    if (q == 0) {
      float z = s + bout[m];
      out[(size_t)(bb0 + row) * M_MOD + m] = 1.f / (1.f + expf(-z));
    }
  }
}

// =====================================================================
extern "C" void kernel_launch(void* const* d_in, const int* in_sizes, int n_in,
                              void* d_out, int out_size, void* d_ws, size_t ws_size,
                              hipStream_t stream) {
  const float* x    = (const float*)d_in[0];
  const float* W0   = (const float*)d_in[1];
  const float* b0   = (const float*)d_in[2];
  const float* W1   = (const float*)d_in[3];
  const float* b1   = (const float*)d_in[4];
  const float* W2   = (const float*)d_in[5];
  const float* b2   = (const float*)d_in[6];
  const float* W3   = (const float*)d_in[7];
  const float* b3   = (const float*)d_in[8];
  const float* W4   = (const float*)d_in[9];
  const float* b4   = (const float*)d_in[10];
  const float* Wout = (const float*)d_in[11];
  const float* bout = (const float*)d_in[12];
  float* out = (float*)d_out;

  // d_ws layout: bf16 hi/lo planes (ushort), ~190 MB total
  unsigned short* p = (unsigned short*)d_ws;
  auto alloc = [&](size_t n) { unsigned short* r = p; p += n; return r; };
  const size_t nX  = (size_t)B_SZ * D_SZ;
  const size_t nW0 = (size_t)N0 * D_SZ;
  const size_t nH0 = (size_t)B_SZ * N0;
  const size_t nW1 = (size_t)M_MOD *  64 *  32;
  const size_t nW2 = (size_t)M_MOD * 128 *  64;
  const size_t nW3 = (size_t)M_MOD * 256 * 128;
  const size_t nW4 = (size_t)M_MOD * 512 * 256;
  unsigned short *Xh = alloc(nX),  *Xl = alloc(nX);
  unsigned short *Wh = alloc(nW0), *Wl = alloc(nW0);
  unsigned short *Hh = alloc(nH0), *Hl = alloc(nH0);
  unsigned short *W1h = alloc(nW1), *W1l = alloc(nW1);
  unsigned short *W2h = alloc(nW2), *W2l = alloc(nW2);
  unsigned short *W3h = alloc(nW3), *W3l = alloc(nW3);
  unsigned short *W4h = alloc(nW4), *W4l = alloc(nW4);

  auto split = [&](const float* s, unsigned short* h, unsigned short* l, size_t n) {
    int n4 = (int)(n / 4);
    split_bf16<<<(n4 + 255) / 256, 256, 0, stream>>>(s, h, l, n4);
  };
  split(x,  Xh,  Xl,  nX);
  split(W0, Wh,  Wl,  nW0);
  split(W1, W1h, W1l, nW1);
  split(W2, W2h, W2l, nW2);
  split(W3, W3h, W3l, nW3);
  split(W4, W4h, W4l, nW4);

  dim3 gA(N0 / TA_N, B_SZ / TA_M);   // 32 x 8
  gemm0_relu_bf16x3<<<gA, 256, 0, stream>>>(Xh, Xl, Wh, Wl, b0, Hh, Hl);

  dim3 gB(M_MOD, B_SZ / BT);         // 128 x 64
  fused_modules<<<gB, 256, 0, stream>>>(Hh, Hl,
                                        W1h, W1l, b1, W2h, W2l, b2,
                                        W3h, W3l, b3, W4h, W4l, b4,
                                        Wout, bout, out);
}
